// InterPart_45827301048588
// MI455X (gfx1250) — matrix operand order repaired and checked by reading, not compiled, not run on hardware
//
#include <hip/hip_runtime.h>
#include <hip/hip_bf16.h>

typedef __bf16 bf16_t;
typedef __attribute__((ext_vector_type(8)))  __bf16 v8bf;
typedef __attribute__((ext_vector_type(16))) __bf16 v16bf;
typedef __attribute__((ext_vector_type(8)))  float  v8f;
typedef __attribute__((ext_vector_type(4)))  unsigned int u32x4;
typedef __attribute__((ext_vector_type(8)))  unsigned int u32x8;

#define NB 32
#define CC 256
#define TT 256
#define VV 25
#define CID 128
#define BBATCH (NB * VV) /* 800 */

// padded LDS row strides (elements) after TDM pad: +4 DWORDs (8 bf16) per row
#define LDP256 264  /* 256-elem rows -> stride 132 DW == 4 mod 64 banks */
#define LDP128 136  /* 128-elem rows -> stride  68 DW == 4 mod 64 banks */

#define WMMA_BF16(a, b, c) \
  __builtin_amdgcn_wmma_f32_16x16x32_bf16(false, (a), false, (b), (short)0, (c), false, false)

// ---- fragment gathers (wave32, 16x16x32 bf16; contiguous k order k = 16*(lane/16) + s in both A and B) ----
__device__ __forceinline__ v16bf frag_a(const bf16_t* __restrict__ base, int lda,
                                        int m0, int k0, int lane) {
  const int hf  = lane >> 4;
  const int row = m0 + (lane & 15);
  const bf16_t* p = base + (size_t)row * lda + (k0 + hf * 16);
  v8bf lo = *(const v8bf*)(p);
  v8bf hi = *(const v8bf*)(p + 8);
  return __builtin_shufflevector(lo, hi, 0, 1, 2, 3, 4, 5, 6, 7, 8, 9, 10, 11, 12, 13, 14, 15);
}

// B fragment from a [N,K] (K-contiguous) global buffer.
__device__ __forceinline__ v16bf frag_b(const bf16_t* __restrict__ base, int ldb,
                                        int n0, int k0, int lane) {
  const int hf  = lane >> 4;
  const int col = n0 + (lane & 15);
  const bf16_t* p = base + (size_t)col * ldb + (k0 + hf * 16);
  v8bf lo = *(const v8bf*)(p);
  v8bf hi = *(const v8bf*)(p + 8);
  return __builtin_shufflevector(lo, hi, 0, 1, 2, 3, 4, 5, 6, 7, 8, 9, 10, 11, 12, 13, 14, 15);
}

// B fragment from an LDS-resident tile with padded row stride -> two ds_load_b128,
// bank-conflict-free (stride == 4 mod 64 banks).
__device__ __forceinline__ v16bf frag_b_lds(const bf16_t* tile, int ldb,
                                            int n0, int k0, int lane) {
  const int hf  = lane >> 4;
  const bf16_t* p = tile + (n0 + (lane & 15)) * ldb + (k0 + hf * 16);
  v8bf lo = *(const v8bf*)(p);
  v8bf hi = *(const v8bf*)(p + 8);
  return __builtin_shufflevector(lo, hi, 0, 1, 2, 3, 4, 5, 6, 7, 8, 9, 10, 11, 12, 13, 14, 15);
}

// ---- Tensor Data Mover: DMA a 2-D bf16 tile (lines x dim0_elems, row stride
// ld_elems) from global memory into LDS with LDS-side padding. D# per CDNA5 ISA
// 8.3/8.4: group0 = count=1 | lds_addr | global_addr[56:0] | type=2; group1 =
// data_size=2B, pad ctrl, tensor/tile dims, tensor_dim0_stride. 2-group form
// (VADDR2/3 = NULL) handles <=2-D tensors. Tracked by TENSORcnt.
//   pad_interval code pi: pad after 2^(pi+1) data DWORDs; pad_amount code pa: pa+1 DWORDs.
__device__ __forceinline__ void tdm_load_tile_2d(void* lds_dst, const void* gsrc,
                                                 unsigned dim0_elems, unsigned lines,
                                                 unsigned ld_elems,
                                                 unsigned pi, unsigned pa) {
  const unsigned long long ga = (unsigned long long)gsrc;
  const unsigned lds_off = (unsigned)(unsigned long long)lds_dst;  // LDS aperture: addr[31:0]
  u32x4 g0 = { 1u,                                   // count=1, is_restore=0
               lds_off,                               // lds_addr
               (unsigned)ga,                          // global_addr[31:0]
               (((unsigned)(ga >> 32)) & 0x01FFFFFFu) | 0x80000000u };  // ga[56:32] | type=2
  u32x8 g1 = { 0x00010000u | (1u << 20) | (pi << 22) | (pa << 25),  // data_size=2B, pad ctrl
               (dim0_elems & 0xFFFFu) << 16,         // tensor_dim0[15:0]
               (lines & 0xFFFFu) << 16,              // tensor_dim0[31:16]=0 | tensor_dim1[15:0]
               (dim0_elems & 0xFFFFu) << 16,         // tensor_dim1[31:16]=0 | tile_dim0
               (lines & 0xFFFFu),                    // tile_dim1 | tile_dim2=0
               ld_elems,                             // tensor_dim0_stride[31:0]
               0u, 0u };
  asm volatile("tensor_load_to_lds %0, %1" :: "s"(g0), "s"(g1) : "memory");
}

// ---------------- pack kernels ----------------
__global__ void k_pack_w(const float* __restrict__ src, bf16_t* __restrict__ dst, int n) {
  int i = blockIdx.x * 256 + threadIdx.x;
  if (i < n) dst[i] = (bf16_t)src[i];
}

// x[N,C,T,V] -> xt[b=(n,v)][t][c] bf16 ; xmt[n][t][c] bf16 (mean over v)
__global__ void k_pack_x(const float* __restrict__ x, bf16_t* __restrict__ xt,
                         bf16_t* __restrict__ xmt) {
  const int idx = blockIdx.x * 256 + threadIdx.x;  // flat (n,c,t)
  const int t = idx & 255;
  const int c = (idx >> 8) & 255;
  const int n = idx >> 16;
  const float* xp = x + (size_t)idx * VV;
  float s = 0.f;
#pragma unroll
  for (int v = 0; v < VV; ++v) {
    const float val = xp[v];
    s += val;
    xt[((size_t)(n * VV + v) * TT + t) * CC + c] = (bf16_t)val;
  }
  xmt[((size_t)n * TT + t) * CC + c] = (bf16_t)(s * (1.0f / 25.0f));
}

// ---------------- GEMM 1: g_x = Wg @ xf + bg -> Gt[n][v*Ci+o][t] ----------------
// grid (T/64, 800), block 256; B tile (64 t-cols x 256 k) staged in LDS by TDM (padded).
__global__ void __launch_bounds__(256) k_gemm_g(const bf16_t* __restrict__ Wg,
                                                const float* __restrict__ bg,
                                                const bf16_t* __restrict__ xt,
                                                bf16_t* __restrict__ Gt) {
  __shared__ bf16_t tile[64 * LDP256];  // ~33 KB
  const int lane = threadIdx.x & 31;
  const int wave = threadIdx.x >> 5;
  const int b = blockIdx.y;
  const int n = b / VV, v = b - n * VV;
  const int m0 = wave * 16;
  const int n0 = blockIdx.x * 64;
  const bf16_t* Bsrc = xt + (size_t)b * TT * CC + (size_t)n0 * CC;
  if (wave == 0) {
    tdm_load_tile_2d(tile, Bsrc, CC, 64, CC, /*pi=*/6u, /*pa=*/3u);
    __builtin_amdgcn_s_wait_tensorcnt(0);
  }
  __syncthreads();
  v8f acc[4] = {};
  for (int k0 = 0; k0 < CC; k0 += 32) {
    v16bf a = frag_a(Wg, CC, m0, k0, lane);
#pragma unroll
    for (int cth = 0; cth < 4; ++cth) {
      v16bf bb = frag_b_lds(tile, LDP256, cth * 16, k0, lane);
      acc[cth] = WMMA_BF16(a, bb, acc[cth]);
    }
  }
  const int hf = lane >> 4, tl = lane & 15;
#pragma unroll
  for (int cth = 0; cth < 4; ++cth) {
    const int t = n0 + cth * 16 + tl;
#pragma unroll
    for (int r = 0; r < 8; ++r) {
      const int o = m0 + r + 8 * hf;
      Gt[((size_t)n * (VV * CID) + v * CID + o) * TT + t] = (bf16_t)(acc[cth][r] + bg[o]);
    }
  }
}

// ---------------- GEMM 2: theta/phi = W @ xmean + b -> [n][t][o] ----------------
__global__ void __launch_bounds__(256) k_gemm_thph(const bf16_t* __restrict__ Wth,
                                                   const float* __restrict__ bth,
                                                   const bf16_t* __restrict__ Wph,
                                                   const float* __restrict__ bph,
                                                   const bf16_t* __restrict__ xmt,
                                                   bf16_t* __restrict__ tht,
                                                   bf16_t* __restrict__ pht) {
  const int lane = threadIdx.x & 31;
  const int wave = threadIdx.x >> 5;
  const int which = blockIdx.z;
  const bf16_t* W = which ? Wph : Wth;
  const float* bias = which ? bph : bth;
  bf16_t* out = which ? pht : tht;
  const int n = blockIdx.y;
  const int m0 = wave * 16;
  const int n0 = blockIdx.x * 64;
  const bf16_t* B = xmt + (size_t)n * TT * CC;
  v8f acc[4] = {};
  for (int k0 = 0; k0 < CC; k0 += 32) {
    v16bf a = frag_a(W, CC, m0, k0, lane);
#pragma unroll
    for (int cth = 0; cth < 4; ++cth) {
      v16bf bb = frag_b(B, CC, n0 + cth * 16, k0, lane);
      acc[cth] = WMMA_BF16(a, bb, acc[cth]);
    }
  }
  const int hf = lane >> 4, tl = lane & 15;
#pragma unroll
  for (int cth = 0; cth < 4; ++cth) {
    const int t = n0 + cth * 16 + tl;
    v8bf vv;
#pragma unroll
    for (int r = 0; r < 8; ++r) vv[r] = (bf16_t)(acc[cth][r] + bias[m0 + r + 8 * hf]);
    *(v8bf*)(out + ((size_t)n * TT + t) * CID + m0 + 8 * hf) = vv;  // packed b128 store
  }
}

// ---------------- GEMM 3 + softmax: f = theta @ phi, row-softmax, bf16 out ----------------
#define SFP 257  /* padded row stride for the fp32 score band */
__global__ void __launch_bounds__(256) k_f_softmax(const bf16_t* __restrict__ tht,
                                                   const bf16_t* __restrict__ pht,
                                                   bf16_t* __restrict__ Fb) {
  __shared__ float sf[16 * SFP];
  __shared__ float sred[16][17];
  const int lane = threadIdx.x & 31;
  const int wave = threadIdx.x >> 5;
  const int n = blockIdx.y;
  const int i0 = blockIdx.x * 16;
  const bf16_t* A = tht + (size_t)n * TT * CID;  // [i][o]
  const bf16_t* B = pht + (size_t)n * TT * CID;  // [j][o]
  const int jb = wave * 32;
  v8f acc[2] = {};
  for (int k0 = 0; k0 < CID; k0 += 32) {
    v16bf a = frag_a(A, CID, i0, k0, lane);
#pragma unroll
    for (int cth = 0; cth < 2; ++cth) {
      v16bf bb = frag_b(B, CID, jb + cth * 16, k0, lane);
      acc[cth] = WMMA_BF16(a, bb, acc[cth]);
    }
  }
  const int hf = lane >> 4, tl = lane & 15;
#pragma unroll
  for (int cth = 0; cth < 2; ++cth)
#pragma unroll
    for (int r = 0; r < 8; ++r) sf[(r + 8 * hf) * SFP + jb + cth * 16 + tl] = acc[cth][r];
  __syncthreads();

  const int row = threadIdx.x >> 4;  // 0..15
  const int ch = threadIdx.x & 15;   // 16-element chunk per thread
  float* rp = &sf[row * SFP + ch * 16];
  float mx = -3.4e38f;
#pragma unroll
  for (int e = 0; e < 16; ++e) mx = fmaxf(mx, rp[e]);
  sred[row][ch] = mx;
  __syncthreads();
  if (ch == 0) {
    float m2 = sred[row][0];
#pragma unroll
    for (int j = 1; j < 16; ++j) m2 = fmaxf(m2, sred[row][j]);
    sred[row][16] = m2;
  }
  __syncthreads();
  const float rowmax = sred[row][16];
  float sum = 0.f;
#pragma unroll
  for (int e = 0; e < 16; ++e) {
    const float ex = __expf(rp[e] - rowmax);
    rp[e] = ex;
    sum += ex;
  }
  __syncthreads();
  sred[row][ch] = sum;
  __syncthreads();
  if (ch == 0) {
    float s2 = sred[row][0];
#pragma unroll
    for (int j = 1; j < 16; ++j) s2 += sred[row][j];
    sred[row][16] = s2;
  }
  __syncthreads();
  const float inv = 1.0f / sred[row][16];
  bf16_t* op = Fb + ((size_t)n * TT + i0 + row) * TT + ch * 16;
#pragma unroll
  for (int e = 0; e < 16; ++e) op[e] = (bf16_t)(rp[e] * inv);
}

// ---------------- GEMM 4: Y = F @ G -> Yt[b][t][o]; B tile via TDM (padded) ----------------
// grid (3200/64, T/128, N), block 256
__global__ void __launch_bounds__(256) k_gemm_y(const bf16_t* __restrict__ Fb,
                                                const bf16_t* __restrict__ Gt,
                                                bf16_t* __restrict__ Yt) {
  __shared__ bf16_t tile[64 * LDP256];  // ~33 KB: 64 j-rows x 256 s (padded)
  const int lane = threadIdx.x & 31;
  const int wave = threadIdx.x >> 5;
  const int n = blockIdx.z;
  const int m0 = blockIdx.y * 128 + wave * 16;  // t rows
  const int n0 = blockIdx.x * 64;               // j cols
  const bf16_t* A = Fb + (size_t)n * TT * TT;                          // [t][s]
  const bf16_t* Bsrc = Gt + (size_t)n * (VV * CID) * TT + (size_t)n0 * TT;  // [j][s]
  if (wave == 0) {
    tdm_load_tile_2d(tile, Bsrc, TT, 64, TT, /*pi=*/6u, /*pa=*/3u);
    __builtin_amdgcn_s_wait_tensorcnt(0);
  }
  __syncthreads();
  v8f acc[4] = {};
  for (int k0 = 0; k0 < TT; k0 += 32) {
    if (k0 + 32 < TT)
      __builtin_prefetch((const void*)(A + (size_t)(m0 + (lane & 15)) * TT + k0 + 32), 0, 3);
    v16bf a = frag_a(A, TT, m0, k0, lane);
#pragma unroll
    for (int cth = 0; cth < 4; ++cth) {
      v16bf bb = frag_b_lds(tile, LDP256, cth * 16, k0, lane);
      acc[cth] = WMMA_BF16(a, bb, acc[cth]);
    }
  }
  const int hf = lane >> 4, tl = lane & 15;
#pragma unroll
  for (int cth = 0; cth < 4; ++cth) {
    const int j = n0 + cth * 16 + tl;
    const int v = j >> 7, o = j & 127;
#pragma unroll
    for (int r = 0; r < 8; ++r) {
      const int t = m0 + r + 8 * hf;
      Yt[((size_t)(n * VV + v) * TT + t) * CID + o] = (bf16_t)acc[cth][r];
    }
  }
}

// ---------------- GEMM 5: wy = Ww @ y + bw (fp32 out); B tile via TDM (padded) ----------------
// grid (T/64, C/128, 800), block 256
__global__ void __launch_bounds__(256) k_gemm_w(const bf16_t* __restrict__ Ww,
                                                const float* __restrict__ bw,
                                                const bf16_t* __restrict__ Yt,
                                                float* __restrict__ wy) {
  __shared__ bf16_t tile[64 * LDP128];  // ~17 KB: 64 t-rows x 128 o (padded)
  const int lane = threadIdx.x & 31;
  const int wave = threadIdx.x >> 5;
  const int b = blockIdx.z;
  const int m0 = blockIdx.y * 128 + wave * 16;  // channel
  const int n0 = blockIdx.x * 64;               // t
  const bf16_t* Bsrc = Yt + (size_t)b * TT * CID + (size_t)n0 * CID;  // [t][o]
  if (wave == 0) {
    tdm_load_tile_2d(tile, Bsrc, CID, 64, CID, /*pi=*/5u, /*pa=*/3u);
    __builtin_amdgcn_s_wait_tensorcnt(0);
  }
  __syncthreads();
  v8f acc[4] = {};
  for (int k0 = 0; k0 < CID; k0 += 32) {
    v16bf a = frag_a(Ww, CID, m0, k0, lane);
#pragma unroll
    for (int cth = 0; cth < 4; ++cth) {
      v16bf bb = frag_b_lds(tile, LDP128, cth * 16, k0, lane);
      acc[cth] = WMMA_BF16(a, bb, acc[cth]);
    }
  }
  const int hf = lane >> 4, tl = lane & 15;
#pragma unroll
  for (int cth = 0; cth < 4; ++cth) {
    const int t = n0 + cth * 16 + tl;
#pragma unroll
    for (int r = 0; r < 8; ++r) {
      const int ch = m0 + r + 8 * hf;
      wy[((size_t)b * CC + ch) * TT + t] = acc[cth][r] + bw[ch];  // coalesced over lanes
    }
  }
}

// ---------------- BN stats: per-channel mean / rsqrt(var+eps) over (b,t) ----------------
__global__ void __launch_bounds__(256) k_bnstats(const float* __restrict__ wy,
                                                 float* __restrict__ mu,
                                                 float* __restrict__ rsig) {
  __shared__ float s1[256], s2[256];
  const int c = blockIdx.x;
  const int tid = threadIdx.x;
  float sum = 0.f, sq = 0.f;
  for (int i = tid; i < BBATCH * TT; i += 256) {
    const int b = i >> 8, t = i & 255;
    const float v = wy[((size_t)b * CC + c) * TT + t];
    sum += v;
    sq += v * v;
  }
  s1[tid] = sum;
  s2[tid] = sq;
  __syncthreads();
  for (int st = 128; st > 0; st >>= 1) {
    if (tid < st) {
      s1[tid] += s1[tid + st];
      s2[tid] += s2[tid + st];
    }
    __syncthreads();
  }
  if (tid == 0) {
    const float inv = 1.0f / (float)(BBATCH * TT);
    const float m = s1[0] * inv;
    const float var = s2[0] * inv - m * m;
    mu[c] = m;
    rsig[c] = rsqrtf(var + 1e-5f);
  }
}

// ---------------- final: normalize + affine + residual + restore [N,C,T,V] ----------------
__global__ void k_final(const float* __restrict__ wy, const float* __restrict__ x,
                        const float* __restrict__ gamma, const float* __restrict__ beta,
                        const float* __restrict__ mu, const float* __restrict__ rsig,
                        float* __restrict__ out) {
  const int idx = blockIdx.x * 256 + threadIdx.x;  // flat (n,c,t)
  const int t = idx & 255;
  const int c = (idx >> 8) & 255;
  const int n = idx >> 16;
  const float g = gamma[c] * rsig[c];
  const float m = mu[c];
  const float be = beta[c];
  const float* xp = x + (size_t)idx * VV;
  float* op = out + (size_t)idx * VV;
#pragma unroll
  for (int v = 0; v < VV; ++v) {
    const float w = wy[((size_t)(n * VV + v) * CC + c) * TT + t];
    op[v] = (w - m) * g + be + xp[v];
  }
}

extern "C" void kernel_launch(void* const* d_in, const int* in_sizes, int n_in,
                              void* d_out, int out_size, void* d_ws, size_t ws_size,
                              hipStream_t stream) {
  (void)in_sizes; (void)n_in; (void)out_size; (void)ws_size;
  const float* x     = (const float*)d_in[0];
  const float* Wg    = (const float*)d_in[1];
  const float* bg    = (const float*)d_in[2];
  const float* Wth   = (const float*)d_in[3];
  const float* bth   = (const float*)d_in[4];
  const float* Wph   = (const float*)d_in[5];
  const float* bph   = (const float*)d_in[6];
  const float* Ww    = (const float*)d_in[7];
  const float* bw    = (const float*)d_in[8];
  const float* gamma = (const float*)d_in[9];
  const float* beta  = (const float*)d_in[10];
  float* out = (float*)d_out;

  char* ws = (char*)d_ws;
  size_t off = 0;
  auto alloc = [&](size_t bytes) -> void* {
    void* p = ws + off;
    off = (off + bytes + 255) & ~(size_t)255;
    return p;
  };
  bf16_t* WgB  = (bf16_t*)alloc((size_t)CID * CC * 2);
  bf16_t* WthB = (bf16_t*)alloc((size_t)CID * CC * 2);
  bf16_t* WphB = (bf16_t*)alloc((size_t)CID * CC * 2);
  bf16_t* WwB  = (bf16_t*)alloc((size_t)CC * CID * 2);
  bf16_t* xt   = (bf16_t*)alloc((size_t)BBATCH * TT * CC * 2);      // ~100 MB
  bf16_t* xmt  = (bf16_t*)alloc((size_t)NB * TT * CC * 2);          // 4 MB
  bf16_t* Gt   = (bf16_t*)alloc((size_t)NB * VV * CID * TT * 2);    // ~50 MB
  bf16_t* tht  = (bf16_t*)alloc((size_t)NB * TT * CID * 2);         // 2 MB
  bf16_t* pht  = (bf16_t*)alloc((size_t)NB * TT * CID * 2);         // 2 MB
  bf16_t* Fb   = (bf16_t*)alloc((size_t)NB * TT * TT * 2);          // 4 MB
  bf16_t* Yt   = (bf16_t*)alloc((size_t)BBATCH * TT * CID * 2);     // ~50 MB
  float*  wy   = (float*)alloc((size_t)BBATCH * CC * TT * 4);       // ~200 MB
  float*  mu   = (float*)alloc(CC * 4);
  float*  rsig = (float*)alloc(CC * 4);

  // 0) weight + activation packing (fp32 -> bf16, K-contiguous layouts)
  k_pack_w<<<128, 256, 0, stream>>>(Wg, WgB, CID * CC);
  k_pack_w<<<128, 256, 0, stream>>>(Wth, WthB, CID * CC);
  k_pack_w<<<128, 256, 0, stream>>>(Wph, WphB, CID * CC);
  k_pack_w<<<128, 256, 0, stream>>>(Ww, WwB, CC * CID);
  k_pack_x<<<(NB * CC * TT) / 256, 256, 0, stream>>>(x, xt, xmt);

  // 1) g_x GEMM  (800 x [128x256 . 256x256]), B tile via padded TDM -> LDS
  k_gemm_g<<<dim3(TT / 64, BBATCH), 256, 0, stream>>>(WgB, bg, xt, Gt);
  // 2) theta & phi GEMMs
  k_gemm_thph<<<dim3(TT / 64, NB, 2), 256, 0, stream>>>(WthB, bth, WphB, bph, xmt, tht, pht);
  // 3) attention scores + fused softmax
  k_f_softmax<<<dim3(TT / 16, NB), 256, 0, stream>>>(tht, pht, Fb);
  // 4) y = f @ g, B tile via padded TDM -> LDS
  k_gemm_y<<<dim3((VV * CID) / 64, TT / 128, NB), 256, 0, stream>>>(Fb, Gt, Yt);
  // 5) wy = Ww @ y + bw, B tile via padded TDM -> LDS
  k_gemm_w<<<dim3(TT / 64, CC / 128, BBATCH), 256, 0, stream>>>(WwB, bw, Yt, wy);
  // 6) BN batch stats
  k_bnstats<<<CC, 256, 0, stream>>>(wy, mu, rsig);
  // 7) normalize + residual + restore [N,C,T,V]
  k_final<<<(NB * CC * TT) / 256, 256, 0, stream>>>(wy, x, gamma, beta, mu, rsig, out);
}
